// SelfReg_8735963480633
// MI455X (gfx1250) — matrix operand order repaired and checked by reading, not compiled, not run on hardware
//
#include <hip/hip_runtime.h>

// ---------------------------------------------------------------------------
// SelfReg forward loss on MI455X (gfx1250, wave32, WMMA).
// GEMMs: v_wmma_f32_16x16x32_bf16 with LDS tiles staged by the CDNA5 async
// global->LDS DMA path (global_load_async_to_lds_b128 / s_wait_asynccnt),
// double-buffered so the next K-slice streams in during the current WMMAs.
// ---------------------------------------------------------------------------

typedef unsigned short u16t;
typedef unsigned int   u32t;

typedef __attribute__((ext_vector_type(16))) __bf16 v16bf;
typedef __attribute__((ext_vector_type(8)))  float  v8f;

union FragAB {         // 16 bf16 = 32 bytes = two b128 loads
  uint4 u4[2];
  v16bf v;
};

__device__ __forceinline__ u16t f2bf(float f) {
  u32t u = __float_as_uint(f);
  u32t r = u + 0x7FFFu + ((u >> 16) & 1u);   // round-to-nearest-even
  return (u16t)(r >> 16);
}

// CDNA5 async copy: 16 bytes per lane, global -> LDS, tracked by ASYNCcnt.
__device__ __forceinline__ void async_ld16(unsigned lds_off, const void* g) {
  asm volatile("global_load_async_to_lds_b128 %0, %1, off"
               :: "v"(lds_off), "v"((unsigned long long)(size_t)g)
               : "memory");
}
__device__ __forceinline__ void wait_async_le6() {
#if __has_builtin(__builtin_amdgcn_s_wait_asynccnt)
  __builtin_amdgcn_s_wait_asynccnt(6);
#else
  asm volatile("s_wait_asynccnt 0x6" ::: "memory");
#endif
}

// ---------------------------------------------------------------------------
// float -> bf16 elementwise convert
// ---------------------------------------------------------------------------
__global__ __launch_bounds__(256) void cvt_bf16_kernel(
    const float* __restrict__ in, u16t* __restrict__ out, int n) {
  for (int i = blockIdx.x * 256 + threadIdx.x; i < n; i += gridDim.x * 256)
    out[i] = f2bf(in[i]);
}

// ---------------------------------------------------------------------------
// W[K][Nsrc] (fp32, row-major) -> WT[Npad][K] (bf16), zero pad for n>=Nsrc.
// ---------------------------------------------------------------------------
__global__ __launch_bounds__(256) void transpose_bf16_kernel(
    const float* __restrict__ W, u16t* __restrict__ WT,
    int K, int Nsrc, int Npad) {
  __shared__ float tile[32][33];
  const int kb = blockIdx.x * 32;
  const int nb = blockIdx.y * 32;
  const int tx = threadIdx.x;     // 0..31
  const int ty = threadIdx.y;     // 0..7
  #pragma unroll
  for (int i = 0; i < 32; i += 8) {
    int k = kb + ty + i, n = nb + tx;
    tile[ty + i][tx] = (k < K && n < Nsrc) ? W[(size_t)k * Nsrc + n] : 0.0f;
  }
  __syncthreads();
  #pragma unroll
  for (int i = 0; i < 32; i += 8) {
    int n = nb + ty + i, k = kb + tx;
    if (n < Npad && k < K) WT[(size_t)n * K + k] = f2bf(tile[tx][ty + i]);
  }
}

// ---------------------------------------------------------------------------
// GEMM: Cout[M][ldc] = A[M][K](bf16) * WT[N][K]^T (bf16), fp32 accumulate.
// Block = 8 waves, block tile 128(M) x 256(N); each wave 64x64 = 4x4 WMMA
// tiles. K-slices of 32 staged to LDS via async DMA, double-buffered:
//   A tile 128x32 (8KB) + B tile 256x32 (16KB), x2 buffers = 48KB LDS.
// Per K-step each thread issues 6 async b128 copies; fragments then come
// from LDS (ds_load_b128) with the documented 16-bit A/B VGPR layouts:
//   A: lanes 0-15 K{0..7,16..23}, lanes 16-31 K{8..15,24..31}
//   B: lanes 0-15 K{0..7,16..23}, lanes 16-31 K{8..15,24..31}
// M,N multiples of 128/256, K multiple of 32 (padded by caller).
// ---------------------------------------------------------------------------
__global__ __launch_bounds__(256) void gemm_bf16_wmma_kernel(
    const u16t* __restrict__ A, const u16t* __restrict__ Bt,
    float* __restrict__ Cout, int M, int N, int K, int ldc) {
  __shared__ alignas(16) u16t shA[2][128 * 32];
  __shared__ alignas(16) u16t shB[2][256 * 32];

  const int tid  = threadIdx.x;
  const int wid  = tid >> 5;
  const int lane = tid & 31;
  const int half = lane >> 4;     // which 16-lane half
  const int l16  = lane & 15;
  const int m_blk = blockIdx.y * 128;
  const int n_blk = blockIdx.x * 256;
  const int m_loc = (wid >> 2) * 64;
  const int n_loc = (wid & 3) * 64;

  // stage one K-slice (k0) of the block tiles into LDS buffer bufi
  auto stage = [&](int bufi, int k0) {
    const unsigned aBase = (unsigned)(size_t)(&shA[bufi][0]);
    const unsigned bBase = (unsigned)(size_t)(&shB[bufi][0]);
    #pragma unroll
    for (int u = 0; u < 2; ++u) {              // A: 512 chunks of 16B
      int cc = tid + u * 256;
      int row = cc >> 2, part = cc & 3;
      async_ld16(aBase + cc * 16,
                 (const void*)(A + (size_t)(m_blk + row) * K + k0 + part * 8));
    }
    #pragma unroll
    for (int u = 0; u < 4; ++u) {              // B: 1024 chunks of 16B
      int cc = tid + u * 256;
      int row = cc >> 2, part = cc & 3;
      async_ld16(bBase + cc * 16,
                 (const void*)(Bt + (size_t)(n_blk + row) * K + k0 + part * 8));
    }
  };

  v8f acc[4][4];
  #pragma unroll
  for (int i = 0; i < 4; ++i)
    #pragma unroll
    for (int j = 0; j < 4; ++j)
      acc[i][j] = (v8f){0.f, 0.f, 0.f, 0.f, 0.f, 0.f, 0.f, 0.f};

  const int niter = K >> 5;
  stage(0, 0);
  int buf = 0;
  for (int it = 0; it < niter; ++it) {
    // prefetch next slice (dummy re-stage of k=0 on the last iteration keeps
    // the ASYNCcnt wait value uniform; its data is never read)
    const int kn = (it + 1 < niter) ? ((it + 1) << 5) : 0;
    stage(buf ^ 1, kn);
    wait_async_le6();          // our 6 copies for 'buf' have landed
    __syncthreads();           // everyone's have

    const uint4* At4 = (const uint4*)(&shA[buf][0]);
    const uint4* Bt4 = (const uint4*)(&shB[buf][0]);
    FragAB a[4], b[4];
    #pragma unroll
    for (int i = 0; i < 4; ++i) {
      const int r4 = (m_loc + i * 16 + l16) * 4;
      a[i].u4[0] = At4[r4 + half];
      a[i].u4[1] = At4[r4 + 2 + half];
    }
    #pragma unroll
    for (int j = 0; j < 4; ++j) {
      const int r4 = (n_loc + j * 16 + l16) * 4;
      b[j].u4[0] = Bt4[r4 + half];
      b[j].u4[1] = Bt4[r4 + 2 + half];
    }
    #pragma unroll
    for (int i = 0; i < 4; ++i)
      #pragma unroll
      for (int j = 0; j < 4; ++j)
        acc[i][j] = __builtin_amdgcn_wmma_f32_16x16x32_bf16(
            false, a[i].v, false, b[j].v, (short)0, acc[i][j], false, false);

    __syncthreads();           // all reads of 'buf' done before next overwrite
    buf ^= 1;
  }

  // C/D layout: VGPR v -> row = half*8 + v, lane l16 -> col.
  #pragma unroll
  for (int i = 0; i < 4; ++i) {
    #pragma unroll
    for (int j = 0; j < 4; ++j) {
      const int row0 = m_blk + m_loc + i * 16 + half * 8;
      const int col  = n_blk + n_loc + j * 16 + l16;
      #pragma unroll
      for (int v = 0; v < 8; ++v)
        Cout[(size_t)(row0 + v) * ldc + col] = acc[i][j][v];
    }
  }
}

// ---------------------------------------------------------------------------
// feat = Z + bf (per column), converted to bf16 for the next GEMMs.
// ---------------------------------------------------------------------------
__global__ __launch_bounds__(256) void feat_bias_bf16_kernel(
    const float* __restrict__ Z, const float* __restrict__ bias,
    u16t* __restrict__ out, int rows, int cols) {
  const int total = rows * cols;
  for (int i = blockIdx.x * 256 + threadIdx.x; i < total; i += gridDim.x * 256)
    out[i] = f2bf(Z[i] + bias[i % cols]);
}

// logits[:, :C] += bc  (padded buffer of stride ldc)
__global__ __launch_bounds__(256) void logits_bias_kernel(
    float* __restrict__ L, const float* __restrict__ bc,
    int rows, int C, int ldc) {
  const int total = rows * C;
  for (int i = blockIdx.x * 256 + threadIdx.x; i < total; i += gridDim.x * 256) {
    int r = i / C, c = i % C;
    L[(size_t)r * ldc + c] += bc[c];
  }
}

// ---------------------------------------------------------------------------
// Per-column biased batch stats. grid.x = cols/64; 4 row-groups per column.
// (Layer bias cancels inside BN's (x - mean), so it is never added.)
// ---------------------------------------------------------------------------
__global__ __launch_bounds__(256) void bn_stats_kernel(
    const float* __restrict__ Z, float* __restrict__ mean,
    float* __restrict__ rstd, int rows, int cols) {
  const int c = blockIdx.x * 64 + (threadIdx.x & 63);
  const int g = threadIdx.x >> 6;   // 0..3
  float s = 0.f, ss = 0.f;
  for (int r = g; r < rows; r += 4) {
    float x = Z[(size_t)r * cols + c];
    s += x; ss += x * x;
  }
  __shared__ float sh_s[256], sh_q[256];
  sh_s[threadIdx.x] = s; sh_q[threadIdx.x] = ss;
  __syncthreads();
  if (g == 0) {
    int t = threadIdx.x;
    s  = sh_s[t] + sh_s[t + 64] + sh_s[t + 128] + sh_s[t + 192];
    ss = sh_q[t] + sh_q[t + 64] + sh_q[t + 128] + sh_q[t + 192];
    float m = s / (float)rows;
    float v = ss / (float)rows - m * m;
    mean[c] = m;
    rstd[c] = rsqrtf(v + 1e-5f);
  }
}

// normalize (+optional ReLU); write bf16 (next GEMM) and/or fp32 (proj).
__global__ __launch_bounds__(256) void bn_apply_kernel(
    const float* __restrict__ Z, const float* __restrict__ mean,
    const float* __restrict__ rstd, const float* __restrict__ gam,
    const float* __restrict__ bet, u16t* __restrict__ out_bf,
    float* __restrict__ out_f, int rows, int cols, int do_relu) {
  const int total = rows * cols;
  for (int i = blockIdx.x * 256 + threadIdx.x; i < total; i += gridDim.x * 256) {
    int c = i % cols;
    float h = gam[c] * (Z[i] - mean[c]) * rstd[c] + bet[c];
    if (do_relu) h = fmaxf(h, 0.f);
    if (out_bf) out_bf[i] = f2bf(h);
    if (out_f)  out_f[i]  = h;
  }
}

// ---------------------------------------------------------------------------
// Per-row logit losses + exact cross entropy. Partials per row (no atomics).
// part[3i] = sum (o - o[s1])^2 ; part[3i+1] = sum (o - mix)^2 ; part[3i+2] = cl_i
// ---------------------------------------------------------------------------
__global__ __launch_bounds__(128) void loss_logits_kernel(
    const float* __restrict__ L, const int* __restrict__ y,
    const int* __restrict__ s1, const int* __restrict__ s2,
    const float* __restrict__ lamp, float* __restrict__ part,
    int C, int ldc) {
  const int i = blockIdx.x;
  const int tid = threadIdx.x;
  const float lam = lamp[0];
  const float* o  = L + (size_t)i * ldc;
  const float* o2 = L + (size_t)s1[i] * ldc;
  const float* o3 = L + (size_t)s2[i] * ldc;

  float d1 = 0.f, d2 = 0.f, mx = -3.402823466e+38f;
  for (int c = tid; c < C; c += 128) {
    float a = o[c], b = o2[c], e = o3[c];
    float m = lam * b + (1.f - lam) * e;
    float t1 = a - b; d1 += t1 * t1;
    float t2 = a - m; d2 += t2 * t2;
    mx = fmaxf(mx, a);
  }
  __shared__ float r1[128], r2[128], r3[128];
  r1[tid] = d1; r2[tid] = d2; r3[tid] = mx;
  __syncthreads();
  for (int s = 64; s > 0; s >>= 1) {
    if (tid < s) {
      r1[tid] += r1[tid + s];
      r2[tid] += r2[tid + s];
      r3[tid] = fmaxf(r3[tid], r3[tid + s]);
    }
    __syncthreads();
  }
  const float rowmax = r3[0];
  __syncthreads();
  float se = 0.f;
  for (int c = tid; c < C; c += 128) se += expf(o[c] - rowmax);
  r3[tid] = se;
  __syncthreads();
  for (int s = 64; s > 0; s >>= 1) {
    if (tid < s) r3[tid] += r3[tid + s];
    __syncthreads();
  }
  if (tid == 0) {
    float lse = logf(r3[0]) + rowmax;
    part[3 * i + 0] = r1[0];
    part[3 * i + 1] = r2[0];
    part[3 * i + 2] = lse - o[y[i]];
  }
}

// ---------------------------------------------------------------------------
// Feature losses: partials per block (fixed grid -> deterministic).
// ---------------------------------------------------------------------------
__global__ __launch_bounds__(256) void loss_feat_kernel(
    const float* __restrict__ P, const int* __restrict__ s1,
    const int* __restrict__ s2, const float* __restrict__ lamp,
    float* __restrict__ partF, int rows, int cols) {
  const float lam = lamp[0];
  const int total = rows * cols;
  float d1 = 0.f, d2 = 0.f;
  for (int i = blockIdx.x * 256 + threadIdx.x; i < total; i += gridDim.x * 256) {
    int r = i / cols, c = i % cols;
    float p  = P[i];
    float p2 = P[(size_t)s1[r] * cols + c];
    float p3 = P[(size_t)s2[r] * cols + c];
    float m  = lam * p2 + (1.f - lam) * p3;
    float t1 = p - p2; d1 += t1 * t1;
    float t2 = p - m;  d2 += t2 * t2;
  }
  __shared__ float r1[256], r2[256];
  r1[threadIdx.x] = d1; r2[threadIdx.x] = d2;
  __syncthreads();
  for (int s = 128; s > 0; s >>= 1) {
    if (threadIdx.x < s) { r1[threadIdx.x] += r1[threadIdx.x + s];
                           r2[threadIdx.x] += r2[threadIdx.x + s]; }
    __syncthreads();
  }
  if (threadIdx.x == 0) {
    partF[2 * blockIdx.x + 0] = r1[0];
    partF[2 * blockIdx.x + 1] = r2[0];
  }
}

// ---------------------------------------------------------------------------
// Final deterministic combine -> d_out[0].
// ---------------------------------------------------------------------------
__global__ __launch_bounds__(256) void finalize_kernel(
    const float* __restrict__ partL, const float* __restrict__ partF,
    const float* __restrict__ lamp, float* __restrict__ out,
    int nL, int nF, int C, int cols) {
  __shared__ float sh[5][256];
  const int tid = threadIdx.x;
  float a0 = 0.f, a1 = 0.f, a2 = 0.f, b0 = 0.f, b1 = 0.f;
  for (int i = tid; i < nL; i += 256) {
    a0 += partL[3 * i + 0];
    a1 += partL[3 * i + 1];
    a2 += partL[3 * i + 2];
  }
  for (int i = tid; i < nF; i += 256) {
    b0 += partF[2 * i + 0];
    b1 += partF[2 * i + 1];
  }
  sh[0][tid] = a0; sh[1][tid] = a1; sh[2][tid] = a2; sh[3][tid] = b0; sh[4][tid] = b1;
  __syncthreads();
  for (int s = 128; s > 0; s >>= 1) {
    if (tid < s)
      #pragma unroll
      for (int j = 0; j < 5; ++j) sh[j][tid] += sh[j][tid + s];
    __syncthreads();
  }
  if (tid == 0) {
    const float lam = lamp[0];
    const float denL = (float)nL * (float)C;
    const float denF = (float)nL * (float)cols;
    float L_ind_logit = sh[0][0] / denL;
    float L_hdl_logit = sh[1][0] / denL;
    float cl          = sh[2][0] / (float)nL;
    float L_ind_feat  = 0.3f * sh[3][0] / denF;
    float L_hdl_feat  = 0.3f * sh[4][0] / denF;
    float Cs = fminf(cl, 1.0f);
    out[0] = cl + Cs * (lam * (L_ind_logit + L_ind_feat) +
                        (1.f - lam) * (L_hdl_logit + L_hdl_feat));
  }
}

// ---------------------------------------------------------------------------
// Host orchestration
// ---------------------------------------------------------------------------
extern "C" void kernel_launch(void* const* d_in, const int* in_sizes, int n_in,
                              void* d_out, int out_size, void* d_ws, size_t ws_size,
                              hipStream_t stream) {
  (void)in_sizes; (void)n_in; (void)out_size; (void)ws_size;
  const int B = 4096, DIN = 2048, FEAT = 2048, HID = 2048, C = 345, CP = 512;

  const float* all_x = (const float*)d_in[0];
  const int*   all_y = (const int*)d_in[1];
  const int*   s1    = (const int*)d_in[2];
  const int*   s2    = (const int*)d_in[3];
  const float* lam   = (const float*)d_in[4];
  const float* Wf    = (const float*)d_in[5];
  const float* bf_   = (const float*)d_in[6];
  const float* Wc    = (const float*)d_in[7];
  const float* bc    = (const float*)d_in[8];
  const float* W1    = (const float*)d_in[9];
  // b1 (d_in[10]) cancels inside BN
  const float* g1    = (const float*)d_in[11];
  const float* be1   = (const float*)d_in[12];
  const float* W2    = (const float*)d_in[13];
  const float* g2    = (const float*)d_in[15];
  const float* be2   = (const float*)d_in[16];
  const float* W3    = (const float*)d_in[17];
  const float* g3    = (const float*)d_in[19];
  const float* be3   = (const float*)d_in[20];

  // workspace layout
  char* ws = (char*)d_ws;
  auto alloc = [&](size_t bytes) -> void* {
    void* p = (void*)ws;
    ws += (bytes + 255) & ~(size_t)255;
    return p;
  };
  u16t* bufA   = (u16t*)alloc((size_t)B * DIN * 2);     // Xb, later h2b
  u16t* bufB   = (u16t*)alloc((size_t)B * FEAT * 2);    // featb, later h1b
  u16t* WfT    = (u16t*)alloc((size_t)FEAT * DIN * 2);
  u16t* W1T    = (u16t*)alloc((size_t)HID * FEAT * 2);
  u16t* W2T    = (u16t*)alloc((size_t)HID * HID * 2);
  u16t* W3T    = (u16t*)alloc((size_t)FEAT * HID * 2);
  u16t* WcT    = (u16t*)alloc((size_t)CP * FEAT * 2);   // zero-padded classifier
  float* Z     = (float*)alloc((size_t)B * FEAT * 4);   // reused per layer; ends as proj
  float* logit = (float*)alloc((size_t)B * CP * 4);
  float* mean  = (float*)alloc((size_t)HID * 4);
  float* rstd  = (float*)alloc((size_t)HID * 4);
  float* partL = (float*)alloc((size_t)B * 3 * 4);
  float* partF = (float*)alloc((size_t)1024 * 2 * 4);

  const dim3 tblk(32, 8);

  // 1. precision conversion + weight transposes
  cvt_bf16_kernel<<<2048, 256, 0, stream>>>(all_x, bufA, B * DIN);
  transpose_bf16_kernel<<<dim3(DIN / 32, FEAT / 32), tblk, 0, stream>>>(Wf, WfT, DIN, FEAT, FEAT);
  transpose_bf16_kernel<<<dim3(FEAT / 32, HID / 32), tblk, 0, stream>>>(W1, W1T, FEAT, HID, HID);
  transpose_bf16_kernel<<<dim3(HID / 32, HID / 32), tblk, 0, stream>>>(W2, W2T, HID, HID, HID);
  transpose_bf16_kernel<<<dim3(HID / 32, FEAT / 32), tblk, 0, stream>>>(W3, W3T, HID, FEAT, FEAT);
  transpose_bf16_kernel<<<dim3(FEAT / 32, CP / 32), tblk, 0, stream>>>(Wc, WcT, FEAT, C, CP);

  // 2. featurizer: Z = Xb @ Wf ; featb = bf16(Z + bf)
  gemm_bf16_wmma_kernel<<<dim3(FEAT / 256, B / 128), 256, 0, stream>>>(bufA, WfT, Z, B, FEAT, DIN, FEAT);
  feat_bias_bf16_kernel<<<4096, 256, 0, stream>>>(Z, bf_, bufB, B, FEAT);

  // 3. classifier logits (padded N=512) + bias
  gemm_bf16_wmma_kernel<<<dim3(CP / 256, B / 128), 256, 0, stream>>>(bufB, WcT, logit, B, CP, FEAT, CP);
  logits_bias_kernel<<<1024, 256, 0, stream>>>(logit, bc, B, C, CP);

  // 4. MLP layer 1: BN + ReLU -> h1b (overwrites featb after its last use)
  gemm_bf16_wmma_kernel<<<dim3(HID / 256, B / 128), 256, 0, stream>>>(bufB, W1T, Z, B, HID, FEAT, HID);
  bn_stats_kernel<<<HID / 64, 256, 0, stream>>>(Z, mean, rstd, B, HID);
  bn_apply_kernel<<<4096, 256, 0, stream>>>(Z, mean, rstd, g1, be1, bufB, nullptr, B, HID, 1);

  // 5. MLP layer 2: BN + ReLU -> h2b (overwrites Xb)
  gemm_bf16_wmma_kernel<<<dim3(HID / 256, B / 128), 256, 0, stream>>>(bufB, W2T, Z, B, HID, HID, HID);
  bn_stats_kernel<<<HID / 64, 256, 0, stream>>>(Z, mean, rstd, B, HID);
  bn_apply_kernel<<<4096, 256, 0, stream>>>(Z, mean, rstd, g2, be2, bufA, nullptr, B, HID, 1);

  // 6. MLP layer 3: BN (no ReLU), proj kept fp32 in place of Z
  gemm_bf16_wmma_kernel<<<dim3(FEAT / 256, B / 128), 256, 0, stream>>>(bufA, W3T, Z, B, FEAT, HID, FEAT);
  bn_stats_kernel<<<FEAT / 64, 256, 0, stream>>>(Z, mean, rstd, B, FEAT);
  bn_apply_kernel<<<4096, 256, 0, stream>>>(Z, mean, rstd, g3, be3, nullptr, Z, B, FEAT, 0);

  // 7. losses + deterministic combine
  loss_logits_kernel<<<B, 128, 0, stream>>>(logit, all_y, s1, s2, lam, partL, C, CP);
  loss_feat_kernel<<<1024, 256, 0, stream>>>(Z, s1, s2, lam, partF, B, FEAT);
  finalize_kernel<<<1, 256, 0, stream>>>(partL, partF, lam, (float*)d_out, B, 1024, C, FEAT);
}
